// TransformerLayer_15118284882407
// MI455X (gfx1250) — matrix operand order repaired and checked by reading, not compiled, not run on hardware
//
#include <hip/hip_runtime.h>
#include <hip/hip_bf16.h>

typedef __attribute__((ext_vector_type(16))) _Float16 v16h;
typedef __attribute__((ext_vector_type(8)))  _Float16 v8h;
typedef __attribute__((ext_vector_type(8)))  float    v8f;
typedef __attribute__((ext_vector_type(4)))  int      v4i;

#define AS1 __attribute__((address_space(1)))
#define AS3 __attribute__((address_space(3)))

// ---------------------------------------------------------------------------
// CDNA5 async global->LDS 16B copy (ASYNCcnt-tracked)
// ---------------------------------------------------------------------------
__device__ __forceinline__ void async_copy16(const _Float16* g, _Float16* l) {
#if __has_builtin(__builtin_amdgcn_global_load_async_to_lds_b128)
    __builtin_amdgcn_global_load_async_to_lds_b128(
        (AS1 v4i*)(uintptr_t)g, (AS3 v4i*)l, 0, 0);
#else
    unsigned lds_off = (unsigned)(uintptr_t)(AS3 void*)l;
    asm volatile("global_load_async_to_lds_b128 %0, %1, off"
                 :: "v"(lds_off), "v"((unsigned long long)(uintptr_t)g)
                 : "memory");
#endif
}

#if __has_builtin(__builtin_amdgcn_s_wait_asynccnt)
#define WAIT_ASYNCCNT(n) __builtin_amdgcn_s_wait_asynccnt(n)
#else
#define WAIT_ASYNCCNT(n) asm volatile("s_wait_asynccnt %0" ::"n"(n) : "memory")
#endif

// ---------------------------------------------------------------------------
// fp32 -> fp16 elementwise convert
// ---------------------------------------------------------------------------
__global__ __launch_bounds__(256) void cvt_f16_kernel(const float* __restrict__ in,
                                                      _Float16* __restrict__ out,
                                                      long long n) {
    long long i = (long long)blockIdx.x * 256 + threadIdx.x;
    long long stride = (long long)gridDim.x * 256;
    for (; i < n; i += stride) out[i] = (_Float16)in[i];
}

// ---------------------------------------------------------------------------
// fp32 [R][C] -> fp16 [C][R] tiled transpose (weights, done once)
// ---------------------------------------------------------------------------
__global__ __launch_bounds__(256) void cvt_transpose_kernel(const float* __restrict__ in,
                                                            _Float16* __restrict__ out,
                                                            int R, int C) {
    __shared__ _Float16 tile[32][33];
    int bx = blockIdx.x * 32;   // column base in `in`
    int by = blockIdx.y * 32;   // row base in `in`
    int tx = threadIdx.x & 31;
    int ty = threadIdx.x >> 5;  // 0..7
#pragma unroll
    for (int i = 0; i < 32; i += 8) {
        int r = by + ty + i, c = bx + tx;
        tile[ty + i][tx] = (_Float16)in[(size_t)r * C + c];
    }
    __syncthreads();
#pragma unroll
    for (int i = 0; i < 32; i += 8) {
        int r = by + tx;        // becomes column of out
        int c = bx + ty + i;    // becomes row of out
        out[(size_t)c * R + r] = tile[tx][ty + i];
    }
}

// ---------------------------------------------------------------------------
// Generic f16 WMMA GEMM:  C[M,N] = A[M,K] * Bt[N,K]^T  (Bt is N x K row-major)
// Block tile 128x128, BK=32, 256 threads (8 waves), wave tile 32x64.
// Double-buffered LDS fed by async global->LDS; pipeline unrolled by 2 so the
// buffer index is a compile-time constant (clean addrspace folding).
// ---------------------------------------------------------------------------
template <bool BIAS, bool RELU, bool RES, bool OF32, bool OF16, bool OTR>
__global__ __launch_bounds__(256) void gemm_f16_wmma(
    const _Float16* __restrict__ A, const _Float16* __restrict__ Bt,
    const float* __restrict__ bias, const float* __restrict__ res,
    float* __restrict__ Cf, _Float16* __restrict__ Ch, _Float16* __restrict__ Cht,
    int M, int N, int K, float scale, int tchunk) {
    constexpr int BM = 128, BN = 128, BK = 32, LDK = 40;  // pad 32->40 halfs: conflict-free b128 reads
    __shared__ _Float16 As[2][BM * LDK];
    __shared__ _Float16 Bs[2][BN * LDK];

    const int tid  = threadIdx.x;
    const int lane = tid & 31;
    const int wave = tid >> 5;
    const int half = lane >> 4;   // 0 / 1
    const int l16  = lane & 15;
    const int wm   = (wave >> 1) * 32;  // wave row offset (4 rows of waves)
    const int wn   = (wave & 1) * 64;   // wave col offset (2 cols of waves)
    const int bm   = blockIdx.y * BM;
    const int bn   = blockIdx.x * BN;

    // each thread owns two 16B chunks of the 128x32 tile (512 chunks total)
    const int r0 = tid >> 2,          c0 = (tid & 3) * 8;
    const int r1 = (tid + 256) >> 2,  c1 = ((tid + 256) & 3) * 8;
    const int i0 = r0 * LDK + c0,     i1 = r1 * LDK + c1;
    const _Float16* gA0 = A  + (size_t)(bm + r0) * K + c0;
    const _Float16* gA1 = A  + (size_t)(bm + r1) * K + c1;
    const _Float16* gB0 = Bt + (size_t)(bn + r0) * K + c0;
    const _Float16* gB1 = Bt + (size_t)(bn + r1) * K + c1;

    v8f c[2][4];
#pragma unroll
    for (int mi = 0; mi < 2; ++mi)
#pragma unroll
        for (int ni = 0; ni < 4; ++ni) c[mi][ni] = (v8f)(0.0f);

    // buf is a literal at every call site -> LDS addresses fold at compile time
    auto issue_tile = [&](int k0, int bufc) {
        async_copy16(gA0 + k0, &As[bufc][i0]);
        async_copy16(gA1 + k0, &As[bufc][i1]);
        async_copy16(gB0 + k0, &Bs[bufc][i0]);
        async_copy16(gB1 + k0, &Bs[bufc][i1]);
    };
    auto compute_tile = [&](int bufc) {
        const _Float16* as = As[bufc];
        const _Float16* bs = Bs[bufc];
        // A lane layout: row = l16; halves hold K {0..7,16..23} / {8..15,24..31}
        v16h a[2];
#pragma unroll
        for (int mi = 0; mi < 2; ++mi) {
            int r = wm + mi * 16 + l16;
            v8h lo = *(const v8h*)(as + r * LDK + half * 8);
            v8h hi = *(const v8h*)(as + r * LDK + 16 + half * 8);
            v16h f;
#pragma unroll
            for (int e = 0; e < 8; ++e) { f[e] = lo[e]; f[8 + e] = hi[e]; }
            a[mi] = f;
        }
        // B lane layout: col = l16; halves hold K {0..7,16..23} / {8..15,24..31}
        v16h b[4];
#pragma unroll
        for (int ni = 0; ni < 4; ++ni) {
            int r = wn + ni * 16 + l16;
            v8h lo = *(const v8h*)(bs + r * LDK + half * 8);
            v8h hi = *(const v8h*)(bs + r * LDK + 16 + half * 8);
            v16h f;
#pragma unroll
            for (int e = 0; e < 8; ++e) { f[e] = lo[e]; f[8 + e] = hi[e]; }
            b[ni] = f;
        }
#pragma unroll
        for (int mi = 0; mi < 2; ++mi)
#pragma unroll
            for (int ni = 0; ni < 4; ++ni)
                c[mi][ni] = __builtin_amdgcn_wmma_f32_16x16x32_f16(
                    false, a[mi], false, b[ni], (short)0, c[mi][ni], false, false);
    };

    const int nt = K / BK;  // even for all shapes used here
    issue_tile(0, 0);

    for (int t = 0; t < nt; t += 2) {
        if (t + 1 < nt) { issue_tile((t + 1) * BK, 1); WAIT_ASYNCCNT(4); }
        else            { WAIT_ASYNCCNT(0); }
        __syncthreads();            // everyone passed their own async wait
        compute_tile(0);
        __syncthreads();            // buf0 free before it is overwritten

        if (t + 1 < nt) {
            if (t + 2 < nt) { issue_tile((t + 2) * BK, 0); WAIT_ASYNCCNT(4); }
            else            { WAIT_ASYNCCNT(0); }
            __syncthreads();
            compute_tile(1);
            __syncthreads();
        }
    }

    // ---- epilogue: D layout: n = l16, m = 8*half + vgpr ----
#pragma unroll
    for (int mi = 0; mi < 2; ++mi) {
#pragma unroll
        for (int ni = 0; ni < 4; ++ni) {
            int n   = bn + wn + ni * 16 + l16;
            int mlo = bm + wm + mi * 16 + half * 8;
            float bv = 0.0f;
            if constexpr (BIAS) bv = bias[n];
#pragma unroll
            for (int r = 0; r < 8; ++r) {
                int m = mlo + r;
                float v = c[mi][ni][r] * scale + bv;
                if constexpr (RELU) v = fmaxf(v, 0.0f);
                if constexpr (RES)  v += res[(size_t)m * N + n];
                if constexpr (OF32) Cf[(size_t)m * N + n] = v;
                if constexpr (OF16) Ch[(size_t)m * N + n] = (_Float16)v;
                if constexpr (OTR) {
                    int bb = m / tchunk, ss = m - bb * tchunk;
                    Cht[((size_t)bb * N + n) * tchunk + ss] = (_Float16)v;
                }
            }
        }
    }
}

// ---------------------------------------------------------------------------
// row softmax: p[q,:] = softmax(s[q,:] + mask[q,:]) -> f16
// one block (256 threads) per row of length S (=2048 -> 8 elems/thread)
// ---------------------------------------------------------------------------
__global__ __launch_bounds__(256) void softmax_kernel(const float* __restrict__ s,
                                                      const float* __restrict__ mask,
                                                      _Float16* __restrict__ p, int S) {
    __shared__ float red[8];
    const int q = blockIdx.x;
    const int tid = threadIdx.x, lane = tid & 31, wave = tid >> 5;
    const float* row  = s    + (size_t)q * S;
    const float* mrow = mask + (size_t)q * S;

    float vals[8];
    float mx = -3.4e38f;
#pragma unroll
    for (int i = 0; i < 8; ++i) {
        int j = tid + i * 256;
        float v = row[j] + mrow[j];
        vals[i] = v;
        mx = fmaxf(mx, v);
    }
#pragma unroll
    for (int o = 16; o; o >>= 1) mx = fmaxf(mx, __shfl_xor(mx, o, 32));
    if (lane == 0) red[wave] = mx;
    __syncthreads();
    float rm = red[0];
#pragma unroll
    for (int i = 1; i < 8; ++i) rm = fmaxf(rm, red[i]);
    __syncthreads();

    float sum = 0.0f;
#pragma unroll
    for (int i = 0; i < 8; ++i) { vals[i] = __expf(vals[i] - rm); sum += vals[i]; }
#pragma unroll
    for (int o = 16; o; o >>= 1) sum += __shfl_xor(sum, o, 32);
    if (lane == 0) red[wave] = sum;
    __syncthreads();
    float rs = 0.0f;
#pragma unroll
    for (int i = 0; i < 8; ++i) rs += red[i];
    float inv = 1.0f / rs;
#pragma unroll
    for (int i = 0; i < 8; ++i)
        p[(size_t)q * S + tid + i * 256] = (_Float16)(vals[i] * inv);
}

// ---------------------------------------------------------------------------
// in-place row LayerNorm over H (=1024 -> 4 elems/thread, 256 threads/row)
// ---------------------------------------------------------------------------
__global__ __launch_bounds__(256) void layernorm_kernel(float* __restrict__ y,
                                                        const float* __restrict__ gamma,
                                                        const float* __restrict__ beta,
                                                        int Hn, float eps) {
    __shared__ float red[8];
    const int tid = threadIdx.x, lane = tid & 31, wave = tid >> 5;
    float* row = y + (size_t)blockIdx.x * Hn;

    float v[4];
    float s = 0.0f, s2 = 0.0f;
#pragma unroll
    for (int i = 0; i < 4; ++i) {
        int j = tid + i * 256;
        float x = row[j];
        v[i] = x;
        s += x;
        s2 += x * x;
    }
#pragma unroll
    for (int o = 16; o; o >>= 1) s += __shfl_xor(s, o, 32);
    if (lane == 0) red[wave] = s;
    __syncthreads();
    float ts = 0.0f;
#pragma unroll
    for (int i = 0; i < 8; ++i) ts += red[i];
    __syncthreads();
#pragma unroll
    for (int o = 16; o; o >>= 1) s2 += __shfl_xor(s2, o, 32);
    if (lane == 0) red[wave] = s2;
    __syncthreads();
    float ts2 = 0.0f;
#pragma unroll
    for (int i = 0; i < 8; ++i) ts2 += red[i];

    float mu = ts / Hn;
    float var = ts2 / Hn - mu * mu;
    float rstd = rsqrtf(var + eps);
#pragma unroll
    for (int i = 0; i < 4; ++i) {
        int j = tid + i * 256;
        row[j] = (v[i] - mu) * rstd * gamma[j] + beta[j];
    }
}

// ---------------------------------------------------------------------------
extern "C" void kernel_launch(void* const* d_in, const int* in_sizes, int n_in,
                              void* d_out, int out_size, void* d_ws, size_t ws_size,
                              hipStream_t stream) {
    const float* x     = (const float*)d_in[0];
    const float* mask  = (const float*)d_in[1];
    const float* qW    = (const float*)d_in[2];
    const float* qb    = (const float*)d_in[3];
    const float* kW    = (const float*)d_in[4];
    const float* kb    = (const float*)d_in[5];
    const float* vW    = (const float*)d_in[6];
    const float* vb    = (const float*)d_in[7];
    const float* w1    = (const float*)d_in[8];
    const float* b1    = (const float*)d_in[9];
    const float* w2    = (const float*)d_in[10];
    const float* b2    = (const float*)d_in[11];
    const float* gamma = (const float*)d_in[12];
    const float* beta  = (const float*)d_in[13];
    float* out = (float*)d_out;

    constexpr int B = 4, S = 2048, H = 1024, F = 4096;
    constexpr int M = B * S;  // 8192
    constexpr float EPS = 1e-5f;

    // workspace carve-out (256B aligned)
    char* wsp = (char*)d_ws;
    size_t off = 0;
    auto alloc = [&](size_t bytes) -> void* {
        void* p = wsp + off;
        off = (off + bytes + 255) & ~(size_t)255;
        return p;
    };
    _Float16* xh    = (_Float16*)alloc((size_t)M * H * 2);
    _Float16* qWt   = (_Float16*)alloc((size_t)H * H * 2);
    _Float16* kWt   = (_Float16*)alloc((size_t)H * H * 2);
    _Float16* vWt   = (_Float16*)alloc((size_t)H * H * 2);
    _Float16* w1t   = (_Float16*)alloc((size_t)F * H * 2);  // [F][H]
    _Float16* w2t   = (_Float16*)alloc((size_t)H * F * 2);  // [H][F]
    _Float16* qh    = (_Float16*)alloc((size_t)M * H * 2);  // [B][S][H]
    _Float16* kh    = (_Float16*)alloc((size_t)M * H * 2);  // [B][S][H]
    _Float16* vTh   = (_Float16*)alloc((size_t)M * H * 2);  // [B][H][S]
    float*    sbuf  = (float*)   alloc((size_t)S * S * 4);  // per-batch scores (reused)
    _Float16* pbuf  = (_Float16*)alloc((size_t)S * S * 2);  // per-batch probs  (reused)
    float*    attnf = (float*)   alloc((size_t)M * H * 4);  // attn fp32 (residual)
    _Float16* attnh = (_Float16*)alloc((size_t)M * H * 2);  // attn f16 (GEMM A)
    _Float16* hbuf  = (_Float16*)alloc((size_t)M * F * 2);  // relu(lin1) f16

    // 1. x -> f16
    cvt_f16_kernel<<<2048, 256, 0, stream>>>(x, xh, (long long)M * H);

    // 2. transpose weights to [N][K] f16
    {
        dim3 g(H / 32, H / 32);
        cvt_transpose_kernel<<<g, 256, 0, stream>>>(qW, qWt, H, H);
        cvt_transpose_kernel<<<g, 256, 0, stream>>>(kW, kWt, H, H);
        cvt_transpose_kernel<<<g, 256, 0, stream>>>(vW, vWt, H, H);
    }
    cvt_transpose_kernel<<<dim3(F / 32, H / 32), 256, 0, stream>>>(w1, w1t, H, F); // -> [F][H]
    cvt_transpose_kernel<<<dim3(H / 32, F / 32), 256, 0, stream>>>(w2, w2t, F, H); // -> [H][F]

    // 3. Q = x*qW + qb ; K = x*kW + kb   (f16 out, row-major)
    {
        dim3 g(H / 128, M / 128);
        gemm_f16_wmma<true, false, false, false, true, false><<<g, 256, 0, stream>>>(
            xh, qWt, qb, nullptr, nullptr, qh, nullptr, M, H, H, 1.0f, 1);
        gemm_f16_wmma<true, false, false, false, true, false><<<g, 256, 0, stream>>>(
            xh, kWt, kb, nullptr, nullptr, kh, nullptr, M, H, H, 1.0f, 1);
        // V = x*vW + vb, stored transposed per batch: vTh[b][h][s]
        gemm_f16_wmma<true, false, false, false, false, true><<<g, 256, 0, stream>>>(
            xh, vWt, vb, nullptr, nullptr, nullptr, vTh, M, H, H, 1.0f, S);
    }

    // 4. attention per batch (score/prob buffers reused; stream order serializes)
    for (int b = 0; b < B; ++b) {
        const _Float16* qb16 = qh  + (size_t)b * S * H;
        const _Float16* kb16 = kh  + (size_t)b * S * H;
        const _Float16* vb16 = vTh + (size_t)b * H * S;
        // scores = q*k^T / sqrt(H)   (Bt = k rows, naturally N x K)
        gemm_f16_wmma<false, false, false, true, false, false>
            <<<dim3(S / 128, S / 128), 256, 0, stream>>>(
                qb16, kb16, nullptr, nullptr, sbuf, nullptr, nullptr,
                S, S, H, 0.03125f, 1);
        // softmax rows (+mask) -> f16 probs
        softmax_kernel<<<S, 256, 0, stream>>>(sbuf, mask, pbuf, S);
        // attn = probs * v    (Bt = vT rows, N=H x K=S)
        gemm_f16_wmma<false, false, false, true, true, false>
            <<<dim3(H / 128, S / 128), 256, 0, stream>>>(
                pbuf, vb16, nullptr, nullptr,
                attnf + (size_t)b * S * H, attnh + (size_t)b * S * H, nullptr,
                S, H, S, 1.0f, 1);
    }

    // 5. FFN1: h = relu(attn*w1 + b1)  -> f16
    gemm_f16_wmma<true, true, false, false, true, false>
        <<<dim3(F / 128, M / 128), 256, 0, stream>>>(
            attnh, w1t, b1, nullptr, nullptr, hbuf, nullptr, M, F, H, 1.0f, 1);

    // 6. FFN2: y = attn + h*w2 + b2 -> fp32 in d_out
    gemm_f16_wmma<true, false, true, true, false, false>
        <<<dim3(H / 128, M / 128), 256, 0, stream>>>(
            hbuf, w2t, b2, attnf, out, nullptr, nullptr, M, H, F, 1.0f, 1);

    // 7. LayerNorm rows of d_out in place
    layernorm_kernel<<<M, 256, 0, stream>>>(out, gamma, beta, H, EPS);
}
